// MambaEncoder_23948737643100
// MI455X (gfx1250) — matrix operand order repaired and checked by reading, not compiled, not run on hardware
//
#include <hip/hip_runtime.h>
#include <hip/hip_bf16.h>
#include <cstdint>

// ---------------- problem constants (match reference) ----------------
#define BB     8
#define LL     2048
#define INF_   256
#define DD     512
#define DI_    1024
#define DS_    64
#define DTR_   32
#define DEPTH_ 4
#define DCONV_ 4
#define EPS_   1e-6f
#define BLr    (BB * LL)   // 16384 rows

#define LDS_STRIDE 36      // 32 payload f32 + 4 pad f32 (TDM pad_interval=32dw, pad_amount=4dw)

// ---------------- CDNA5 vector types ----------------
typedef __attribute__((ext_vector_type(16))) __bf16 v16bf;
typedef __attribute__((ext_vector_type(8)))  float  v8f;
typedef __attribute__((ext_vector_type(4)))  unsigned int u32x4;
typedef __attribute__((ext_vector_type(8)))  int    i32x8;
typedef __attribute__((ext_vector_type(4)))  int    i32x4;

// ---------------- Tensor Data Mover availability ----------------
#ifdef __HIP_DEVICE_COMPILE__
#if __has_builtin(__builtin_amdgcn_tensor_load_to_lds) && __has_builtin(__builtin_amdgcn_s_wait_tensorcnt)
#define USE_TDM 1
#endif
#endif
#ifndef USE_TDM
#define USE_TDM 0
#endif

#if USE_TDM
// Issue one 2D TDM tile load: global (row-major, stride_elems f32/row) -> LDS.
// Tile = tile_rows x 32 f32. Rows >= tensor_rows are zero-filled by the TDM
// (OOB reads return zero), which handles M/N edge tiles. LDS rows are padded
// to LDS_STRIDE f32 via the descriptor pad feature.
__device__ __forceinline__ void tdm_load_2d(unsigned lds_off, const float* gaddr,
                                            unsigned tile_rows, unsigned tensor_rows,
                                            unsigned stride_elems)
{
    unsigned long long ga = (unsigned long long)(uintptr_t)gaddr;

    u32x4 g0;
    g0[0] = 1u;                                             // count=1 (valid), user mode
    g0[1] = lds_off;                                        // lds_addr (bytes)
    g0[2] = (unsigned)(ga & 0xFFFFFFFFull);                 // global_addr[31:0]
    g0[3] = (unsigned)((ga >> 32) & 0x1FFFFFFull)           // global_addr[56:32]
          | (2u << 30);                                     // type=2 (image)

    i32x8 g1;
    g1[0] = (int)0x07120000u;   // data_size=4B | pad_en | pad_interval=32dw | pad_amount=4dw
    g1[1] = (int)((32u & 0xFFFFu) << 16);                   // tensor_dim0 = 32 (lo 16)
    g1[2] = (int)(((tensor_rows & 0xFFFFu) << 16));         // td0 hi=0 | tensor_dim1 lo
    g1[3] = (int)(((tensor_rows >> 16) & 0xFFFFu)           // tensor_dim1 hi
          | (32u << 16));                                   // tile_dim0 = 32
    g1[4] = (int)(tile_rows & 0xFFFFu);                     // tile_dim1 | tile_dim2=0
    g1[5] = (int)stride_elems;                              // tensor_dim0_stride[31:0]
    g1[6] = 0;                                              // stride hi | dim1_stride lo
    g1[7] = 0;

    i32x4 z4 = {0, 0, 0, 0};
#if __clang_major__ >= 23
    i32x8 z8 = {0, 0, 0, 0, 0, 0, 0, 0};
    __builtin_amdgcn_tensor_load_to_lds(g0, g1, z4, z4, z8, 0);
#else
    __builtin_amdgcn_tensor_load_to_lds(g0, g1, z4, z4, 0);
#endif
}
#endif

// =====================================================================
// Generic bf16-WMMA GEMM:  C[M,N] = A[M,K](lda) * W[N,K](ldw)^T + bias
// mode: 0 = bias only, 1 = bias + softplus
// Block: 256 threads (8 waves), tile 128(M) x 64(N), K-step 32.
// Waves 4(M) x 2(N); each wave owns 32x32 = 4 x v_wmma_f32_16x16x32_bf16.
// Tiles staged global->LDS by the Tensor Data Mover (double buffered,
// s_wait_tensorcnt), f32 in LDS, converted to bf16 at fragment build.
// =====================================================================
__global__ __launch_bounds__(256) void wmma_gemm_bf16(
    const float* __restrict__ A, int lda,
    const float* __restrict__ W, int ldw,
    const float* __restrict__ bias,
    float* __restrict__ C, int ldc,
    int M, int N, int K, int mode)
{
    __shared__ alignas(16) float Ash[2][128][LDS_STRIDE];
    __shared__ alignas(16) float Wsh[2][64][LDS_STRIDE];

    const int tid     = threadIdx.x;
    const int lane    = tid & 31;
    const int waveId  = tid >> 5;
    const int l15     = lane & 15;
    const bool hiHalf = (lane >= 16);

    const int blockN = blockIdx.x * 64;
    const int blockM = blockIdx.y * 128;

    const int wm = (waveId >> 1) * 32;  // 0,32,64,96
    const int wn = (waveId & 1) * 32;   // 0,32

    v8f acc[2][2] = {{{}, {}}, {{}, {}}};

    const int nk = K >> 5;

#if USE_TDM
    if (waveId == 0) {
        tdm_load_2d((unsigned)(uintptr_t)&Ash[0][0][0], A + (size_t)blockM * lda,
                    128u, (unsigned)(M - blockM), (unsigned)lda);
        tdm_load_2d((unsigned)(uintptr_t)&Wsh[0][0][0], W + (size_t)blockN * ldw,
                    64u, (unsigned)(N - blockN), (unsigned)ldw);
    }
#endif

    for (int i = 0; i < nk; ++i) {
        const int kt  = i << 5;
        int cur = 0;

#if USE_TDM
        cur = i & 1;
        if (waveId == 0) {
            if (i + 1 < nk) {   // prefetch next K-tile into the other buffer
                const int nxt = 1 - cur;
                tdm_load_2d((unsigned)(uintptr_t)&Ash[nxt][0][0],
                            A + (size_t)blockM * lda + kt + 32,
                            128u, (unsigned)(M - blockM), (unsigned)lda);
                tdm_load_2d((unsigned)(uintptr_t)&Wsh[nxt][0][0],
                            W + (size_t)blockN * ldw + kt + 32,
                            64u, (unsigned)(N - blockN), (unsigned)ldw);
                __builtin_amdgcn_s_wait_tensorcnt(2);  // current pair done, prefetch in flight
            } else {
                __builtin_amdgcn_s_wait_tensorcnt(0);  // drain
            }
        }
        __syncthreads();
#else
        // ---- fallback: branch-free vectorized staging, f32 -> LDS ----
        {   // A: 128 rows x 32, 16 f32 per thread
            const int r = tid >> 1, c = (tid & 1) << 4;
            const int gr = blockM + r;
            const bool ok = (gr < M);
            const float4* gp = (const float4*)(A + (size_t)(ok ? gr : 0) * lda + kt + c);
            if (ok) __builtin_prefetch(gp + 8, 0, 1);
            float4 v0 = gp[0], v1 = gp[1], v2 = gp[2], v3 = gp[3];
            if (!ok) { v0 = v1 = v2 = v3 = make_float4(0.f, 0.f, 0.f, 0.f); }
            float4* sp = (float4*)&Ash[0][r][c];
            sp[0] = v0; sp[1] = v1; sp[2] = v2; sp[3] = v3;
        }
        {   // W: 64 rows x 32, 8 f32 per thread
            const int r = tid >> 2, c = (tid & 3) << 3;
            const int gr = blockN + r;
            const bool ok = (gr < N);
            const float4* gp = (const float4*)(W + (size_t)(ok ? gr : 0) * ldw + kt + c);
            if (ok) __builtin_prefetch(gp + 8, 0, 1);
            float4 v0 = gp[0], v1 = gp[1];
            if (!ok) { v0 = v1 = make_float4(0.f, 0.f, 0.f, 0.f); }
            float4* sp = (float4*)&Wsh[0][r][c];
            sp[0] = v0; sp[1] = v1;
        }
        __syncthreads();
#endif

        // ---- fragments (f32 LDS -> bf16 regs), per documented layouts ----
        const int kb = hiHalf ? 16 : 0;  // B: same K order as A (lane<16 K 0-7/16-23, else 8-15/24-31)
        const int b0 = hiHalf ? 8 : 0;   // A: lane<16 holds K 0-7/16-23, else 8-15/24-31

        v16bf bfrag[2];
        #pragma unroll
        for (int g = 0; g < 2; ++g) {
            const float* wr = &Wsh[cur][wn + g * 16 + l15][0];
            #pragma unroll
            for (int e = 0; e < 16; ++e) bfrag[g][e] = (__bf16)wr[(e < 8) ? (b0 + e) : (b0 + 8 + e)];
        }
        v16bf afrag[2];
        #pragma unroll
        for (int f = 0; f < 2; ++f) {
            const float* ar = &Ash[cur][wm + f * 16 + l15][0];
            #pragma unroll
            for (int e = 0; e < 8; ++e) {
                afrag[f][e]     = (__bf16)ar[b0 + e];
                afrag[f][e + 8] = (__bf16)ar[b0 + 16 + e];
            }
        }

        #pragma unroll
        for (int f = 0; f < 2; ++f)
            #pragma unroll
            for (int g = 0; g < 2; ++g)
                acc[f][g] = __builtin_amdgcn_wmma_f32_16x16x32_bf16(
                    false, afrag[f], false, bfrag[g], (short)0, acc[f][g], false, false);

        __syncthreads();
    }

    // ---- epilogue: D layout -> VGPR r holds M=r (+8 for hi lanes), N=lane&15
    const int mAdd = hiHalf ? 8 : 0;
    #pragma unroll
    for (int g = 0; g < 2; ++g) {
        const int col = blockN + wn + g * 16 + l15;
        if (col >= N) continue;
        const float bv = bias ? bias[col] : 0.0f;
        #pragma unroll
        for (int f = 0; f < 2; ++f) {
            #pragma unroll
            for (int r = 0; r < 8; ++r) {
                const int row = blockM + wm + f * 16 + r + mAdd;
                if (row >= M) continue;
                float v = acc[f][g][r] + bv;
                if (mode == 1) v = (v > 20.0f) ? v : log1pf(__expf(v));  // softplus
                C[(size_t)row * ldc + col] = v;
            }
        }
    }
}

// =====================================================================
// x = inputs@Wemb^T + b (GEMM) then += pos_emb + cls_table[label]
// =====================================================================
__global__ void embed_add_kernel(float* __restrict__ x,
                                 const float* __restrict__ pos,
                                 const float* __restrict__ cls,
                                 const int* __restrict__ label)
{
    size_t idx = (size_t)blockIdx.x * blockDim.x + threadIdx.x;
    if (idx >= (size_t)BLr * DD) return;
    int d = idx % DD;
    int l = (idx / DD) % LL;
    int b = idx / ((size_t)DD * LL);
    x[idx] += pos[(size_t)l * DD + d] + cls[(size_t)label[b] * DD + d];
}

// =====================================================================
// LayerNorm: one wave32 per row of D=512; 8 rows per 256-thread block
// =====================================================================
__global__ __launch_bounds__(256) void layernorm_kernel(
    const float* __restrict__ x, const float* __restrict__ w,
    const float* __restrict__ bsh, float* __restrict__ y)
{
    const int lane = threadIdx.x & 31;
    const int row  = blockIdx.x * 8 + (threadIdx.x >> 5);
    if (row >= BLr) return;
    const float* xr = x + (size_t)row * DD;

    float v[DD / 32];
    float s1 = 0.0f, s2 = 0.0f;
    #pragma unroll
    for (int j = 0; j < DD / 32; ++j) {
        float t = xr[lane + j * 32];
        v[j] = t; s1 += t; s2 += t * t;
    }
    #pragma unroll
    for (int off = 16; off > 0; off >>= 1) {
        s1 += __shfl_xor(s1, off, 32);
        s2 += __shfl_xor(s2, off, 32);
    }
    const float mu  = s1 * (1.0f / DD);
    const float var = s2 * (1.0f / DD) - mu * mu;
    const float rs  = rsqrtf(var + EPS_);

    float* yr = y + (size_t)row * DD;
    #pragma unroll
    for (int j = 0; j < DD / 32; ++j) {
        int d = lane + j * 32;
        yr[d] = (v[j] - mu) * rs * w[d] + bsh[d];
    }
}

// =====================================================================
// Causal depthwise conv (k=4) over xz[..., :DI] + bias + SiLU -> xp
// =====================================================================
__global__ void conv_silu_kernel(const float* __restrict__ xz,
                                 const float* __restrict__ cw,
                                 const float* __restrict__ cb,
                                 float* __restrict__ xp)
{
    size_t idx = (size_t)blockIdx.x * blockDim.x + threadIdx.x;
    if (idx >= (size_t)BLr * DI_) return;
    int d = idx % DI_;
    int l = (idx / DI_) % LL;
    int b = idx / ((size_t)DI_ * LL);

    float acc = cb[d];
    #pragma unroll
    for (int j = 0; j < DCONV_; ++j) {
        int ll = l - (DCONV_ - 1) + j;
        if (ll >= 0)
            acc += xz[((size_t)b * LL + ll) * (2 * DI_) + d] * cw[d * DCONV_ + j];
    }
    xp[idx] = acc / (1.0f + __expf(-acc));  // silu
}

// =====================================================================
// Selective scan: one thread per (b,d) channel; 64 states in VGPRs.
// B/C rows for each timestep staged through LDS (shared by the block).
// =====================================================================
__global__ __launch_bounds__(256) void scan_kernel(
    const float* __restrict__ dt,    // [B*L, DI]
    const float* __restrict__ xdbl,  // [B*L, 160]  B at +32, C at +96
    const float* __restrict__ xp,    // [B*L, DI]
    const float* __restrict__ alog,  // [DI, DS]
    float* __restrict__ ys)          // [B*L, DI]
{
    __shared__ float sB[DS_];
    __shared__ float sC[DS_];

    const int d = blockIdx.x * blockDim.x + threadIdx.x;  // 0..DI-1
    const int b = blockIdx.y;

    float Arow[DS_];
    #pragma unroll
    for (int s = 0; s < DS_; ++s)
        Arow[s] = -__expf(alog[(size_t)d * DS_ + s]);

    float hs[DS_];
    #pragma unroll
    for (int s = 0; s < DS_; ++s) hs[s] = 0.0f;

    for (int t = 0; t < LL; ++t) {
        const size_t row = (size_t)b * LL + t;
        if (threadIdx.x < DS_)
            sB[threadIdx.x] = xdbl[row * 160 + DTR_ + threadIdx.x];
        else if (threadIdx.x < 2 * DS_)
            sC[threadIdx.x - DS_] = xdbl[row * 160 + DTR_ + DS_ + (threadIdx.x - DS_)];
        __syncthreads();

        const float dtv = dt[row * DI_ + d];
        const float dx  = dtv * xp[row * DI_ + d];
        float acc = 0.0f;
        #pragma unroll
        for (int s = 0; s < DS_; ++s) {
            hs[s] = hs[s] * __expf(dtv * Arow[s]) + dx * sB[s];
            acc += hs[s] * sC[s];
        }
        ys[row * DI_ + d] = acc;
        __syncthreads();
    }
}

// =====================================================================
// y = (ys + xp * D_skip) * silu(z)   (z = xz[..., DI:])  in-place in ys
// =====================================================================
__global__ void gate_kernel(float* __restrict__ ys,
                            const float* __restrict__ xp,
                            const float* __restrict__ xz,
                            const float* __restrict__ dsk)
{
    size_t idx = (size_t)blockIdx.x * blockDim.x + threadIdx.x;
    if (idx >= (size_t)BLr * DI_) return;
    int d = idx % DI_;
    size_t row = idx / DI_;
    float z = xz[row * (2 * DI_) + DI_ + d];
    float y = ys[idx] + xp[idx] * dsk[d];
    ys[idx] = y * (z / (1.0f + __expf(-z)));
}

__global__ void residual_add_kernel(float* __restrict__ x,
                                    const float* __restrict__ o)
{
    size_t idx = (size_t)blockIdx.x * blockDim.x + threadIdx.x;
    if (idx < (size_t)BLr * DD) x[idx] += o[idx];
}

__global__ void extract_last_kernel(const float* __restrict__ x,
                                    float* __restrict__ out)
{
    int i = blockIdx.x * blockDim.x + threadIdx.x;
    if (i >= BB * DD) return;
    int b = i / DD, d = i % DD;
    out[i] = x[((size_t)b * LL + (LL - 1)) * DD + d];
}

// =====================================================================
// Host orchestration
// =====================================================================
static inline void launch_gemm(const float* A, int lda, const float* W, int ldw,
                               const float* bias, float* C, int ldc,
                               int M, int N, int K, int mode, hipStream_t s)
{
    dim3 grid((N + 63) / 64, (M + 127) / 128);
    wmma_gemm_bf16<<<grid, 256, 0, s>>>(A, lda, W, ldw, bias, C, ldc, M, N, K, mode);
}

extern "C" void kernel_launch(void* const* d_in, const int* in_sizes, int n_in,
                              void* d_out, int out_size, void* d_ws, size_t ws_size,
                              hipStream_t stream)
{
    const float* inputs    = (const float*)d_in[0];
    const int*   label     = (const int*)  d_in[1];
    const float* emb_w     = (const float*)d_in[2];
    const float* emb_b     = (const float*)d_in[3];
    const float* pos_emb   = (const float*)d_in[4];
    const float* cls_table = (const float*)d_in[5];
    const float* norm_w    = (const float*)d_in[6];
    const float* norm_b    = (const float*)d_in[7];
    const float* in_proj_w = (const float*)d_in[8];
    const float* in_proj_b = (const float*)d_in[9];
    const float* conv_w    = (const float*)d_in[10];
    const float* conv_b    = (const float*)d_in[11];
    const float* x_proj_w  = (const float*)d_in[12];
    const float* dt_proj_w = (const float*)d_in[13];
    const float* dt_proj_b = (const float*)d_in[14];
    const float* A_log     = (const float*)d_in[15];
    const float* D_skip    = (const float*)d_in[16];
    const float* out_proj_w= (const float*)d_in[17];
    const float* out_proj_b= (const float*)d_in[18];

    float* ws = (float*)d_ws;
    size_t off = 0;
    auto carve = [&](size_t n) { float* p = ws + off; off += n; return p; };

    float* bufX    = carve((size_t)BLr * DD);        //  33.5 MB
    float* bufH    = carve((size_t)BLr * DD);        //  LN out / out_proj out
    float* bufXZ   = carve((size_t)BLr * 2 * DI_);   // 134 MB
    float* bufXP   = carve((size_t)BLr * DI_);       //  67 MB
    float* bufXDBL = carve((size_t)BLr * 160);       //  10.5 MB
    float* bufDT   = carve((size_t)BLr * DI_);       //  67 MB
    float* bufYS   = carve((size_t)BLr * DI_);       //  67 MB
    (void)ws_size; (void)n_in; (void)in_sizes; (void)out_size;

    const size_t nBLD  = (size_t)BLr * DD;
    const size_t nBLDI = (size_t)BLr * DI_;

    // 1) embedding GEMM + positional/class additions
    launch_gemm(inputs, INF_, emb_w, INF_, emb_b, bufX, DD,
                BLr, DD, INF_, 0, stream);
    embed_add_kernel<<<(nBLD + 255) / 256, 256, 0, stream>>>(bufX, pos_emb, cls_table, label);

    for (int i = 0; i < DEPTH_; ++i) {
        // 2) LayerNorm
        layernorm_kernel<<<BLr / 8, 256, 0, stream>>>(
            bufX, norm_w + (size_t)i * DD, norm_b + (size_t)i * DD, bufH);

        // 3) in_proj: [BL,512] x [2048,512]^T -> xz [BL,2048]
        launch_gemm(bufH, DD, in_proj_w + (size_t)i * 2 * DI_ * DD, DD,
                    in_proj_b + (size_t)i * 2 * DI_, bufXZ, 2 * DI_,
                    BLr, 2 * DI_, DD, 0, stream);

        // 4) causal depthwise conv + SiLU -> xp
        conv_silu_kernel<<<(nBLDI + 255) / 256, 256, 0, stream>>>(
            bufXZ, conv_w + (size_t)i * DI_ * DCONV_, conv_b + (size_t)i * DI_, bufXP);

        // 5) x_proj: [BL,1024] x [160,1024]^T -> xdbl (no bias)
        launch_gemm(bufXP, DI_, x_proj_w + (size_t)i * (DTR_ + 2 * DS_) * DI_, DI_,
                    nullptr, bufXDBL, DTR_ + 2 * DS_,
                    BLr, DTR_ + 2 * DS_, DI_, 0, stream);

        // 6) dt_proj on strided slice xdbl[..., :32] + softplus epilogue
        launch_gemm(bufXDBL, DTR_ + 2 * DS_, dt_proj_w + (size_t)i * DI_ * DTR_, DTR_,
                    dt_proj_b + (size_t)i * DI_, bufDT, DI_,
                    BLr, DI_, DTR_, 1, stream);

        // 7) selective scan (sequential over L, state in VGPRs)
        scan_kernel<<<dim3(DI_ / 256, BB), 256, 0, stream>>>(
            bufDT, bufXDBL, bufXP, A_log + (size_t)i * DI_ * DS_, bufYS);

        // 8) y = (ys + xp*D) * silu(z)
        gate_kernel<<<(nBLDI + 255) / 256, 256, 0, stream>>>(
            bufYS, bufXP, bufXZ, D_skip + (size_t)i * DI_);

        // 9) out_proj: [BL,1024] x [512,1024]^T -> bufH, then residual
        launch_gemm(bufYS, DI_, out_proj_w + (size_t)i * DD * DI_, DI_,
                    out_proj_b + (size_t)i * DD, bufH, DD,
                    BLr, DD, DI_, 0, stream);
        residual_add_kernel<<<(nBLD + 255) / 256, 256, 0, stream>>>(bufX, bufH);
    }

    // 10) output = x[:, -1, :]
    extract_last_kernel<<<(BB * DD + 255) / 256, 256, 0, stream>>>(bufX, (float*)d_out);
}
